// Model_35227321761996
// MI455X (gfx1250) — matrix operand order repaired and checked by reading, not compiled, not run on hardware
//
#include <hip/hip_runtime.h>
#include <cstdint>
#include <cstddef>

// ---- model dims (match reference) ----
#define B_      4
#define L_      4096
#define CIN_    7
#define NMARK_  4
#define D_      128
#define E_      256
#define N_      16
#define R_      8
#define K_      4
#define KE_     3
#define NL_     2
#define COUT_   7
#define MROWS   (B_ * L_)        // 16384
#define F_      (R_ + 2 * N_)    // 40
#define S_      64               // scan chunks
#define CL_     (L_ / S_)        // 64 steps per chunk

typedef __bf16 bf16;
typedef __attribute__((ext_vector_type(16))) __bf16 v16bf;
typedef __attribute__((ext_vector_type(8)))  __bf16 v8bf;
typedef __attribute__((ext_vector_type(8)))  float  v8f;

// round-to-nearest-even f32 -> bf16
__device__ __forceinline__ bf16 f2bf(float f) {
  union { float f; uint32_t u; } a; a.f = f;
  uint32_t r = (a.u + 0x7FFFu + ((a.u >> 16) & 1u)) >> 16;
  union { uint16_t s; bf16 b; } o; o.s = (uint16_t)r;
  return o.b;
}

// ---------------- one-shot f32 -> bf16 weight conversion --------------------
__global__ void cvt_bf16_kernel(const float* __restrict__ src, bf16* __restrict__ dst,
                                int n) {
  int i = blockIdx.x * blockDim.x + threadIdx.x;
  if (i < n) dst[i] = f2bf(src[i]);
}

// ============================================================================
// WMMA bf16 GEMM: C[M,N](bf16) = A[M,K](bf16) * Wb[N,K](bf16)^T
// Block: 8 waves. All waves share the 16-row A tile -> wave 0 stages it into
// LDS with async global->LDS copies (ASYNCcnt), then fragments come from LDS.
// Each wave computes TN adjacent 16x16 tiles reusing one A fragment.
// ============================================================================
// A fragment from LDS: lane l<16 -> row l, K={khalf..+7, khalf+16..+23},
// khalf = 0 (l<16) / 8 (l>=16).  (ISA 7.12.2, 16-bit A 16x32)
template <int KD>
__device__ __forceinline__ v16bf a_frag_lds(const bf16* As, int k0) {
  const int lane  = threadIdx.x & 31;
  const int row   = lane & 15;
  const int khalf = (lane < 16) ? 0 : 8;
  const bf16* p = As + row * KD + k0 + khalf;
  v8bf lo = *reinterpret_cast<const v8bf*>(p);        // ds_load_b128
  v8bf hi = *reinterpret_cast<const v8bf*>(p + 16);   // ds_load_b128
  v16bf out;
#pragma unroll
  for (int i = 0; i < 8; ++i) { out[i] = lo[i]; out[8 + i] = hi[i]; }
  return out;
}

// B fragment (bf16 W, row-major NxK): lane l<16 -> col l, K={k0..+7, k0+16..+23};
// lane l>=16 -> col l-16, K={k0+8..+15, k0+24..+31}. 2x b128 loads.
__device__ __forceinline__ v16bf b_frag_bf(const bf16* __restrict__ Wb, int ldw,
                                           int n0, int k0) {
  const int lane  = threadIdx.x & 31;
  const int n     = n0 + (lane & 15);
  const int khalf = (lane < 16) ? 0 : 8;
  const bf16* p   = Wb + (size_t)n * ldw + k0 + khalf;
  v8bf lo = *reinterpret_cast<const v8bf*>(p);        // global_load_b128
  v8bf hi = *reinterpret_cast<const v8bf*>(p + 16);   // global_load_b128
  v16bf out;
#pragma unroll
  for (int i = 0; i < 8; ++i) { out[i] = lo[i]; out[8 + i] = hi[i]; }
  return out;
}

template <int KD, int TN>
__global__ __launch_bounds__(256)
void wmma_gemm_bf16(const bf16* __restrict__ A, const bf16* __restrict__ Wb,
                    bf16* __restrict__ C, int N) {
  __shared__ bf16 As[16 * KD];                       // 4KB (KD=128) / 8KB (KD=256)
  const int tid  = threadIdx.x;
  const int wave = tid >> 5;
  const int lane = tid & 31;
  const int m0    = blockIdx.x * 16;
  const int nbase = (blockIdx.y * 8 + wave) * (16 * TN);

  // ---- wave 0: async-stage the 16 x KD bf16 A tile into LDS ----
  if (wave == 0) {
    const char* gsrc = (const char*)(A + (size_t)m0 * KD);
    const uint32_t lds_base = (uint32_t)(uintptr_t)(&As[0]);
    constexpr int bytes = 16 * KD * 2;
#pragma unroll
    for (int off0 = 0; off0 < bytes; off0 += 512) {   // 32 lanes x 16B
      const int off = off0 + lane * 16;
      uint32_t ldst = lds_base + (uint32_t)off;
      uint64_t gaddr = (uint64_t)(uintptr_t)(gsrc + off);
      asm volatile("global_load_async_to_lds_b128 %0, %1, off"
                   :: "v"(ldst), "v"(gaddr) : "memory");
    }
    asm volatile("s_wait_asynccnt 0x0" ::: "memory");
  }
  __syncthreads();

  v8f acc[TN];
#pragma unroll
  for (int t = 0; t < TN; ++t) acc[t] = (v8f){0.f,0.f,0.f,0.f,0.f,0.f,0.f,0.f};

#pragma unroll
  for (int k0 = 0; k0 < KD; k0 += 32) {              // fully unrolled
    v16bf a = a_frag_lds<KD>(As, k0);
#pragma unroll
    for (int t = 0; t < TN; ++t) {
      v16bf b = b_frag_bf(Wb, KD, nbase + 16 * t, k0);
      acc[t] = __builtin_amdgcn_wmma_f32_16x16x32_bf16(false, a, false, b,
                                                       (short)0, acc[t], false, false);
    }
  }
  // C/D layout: VGPR r -> (M = r | r+8 per half-wave, N = lane&15)
  const int rbase = m0 + ((lane < 16) ? 0 : 8);
#pragma unroll
  for (int t = 0; t < TN; ++t) {
    const int col = nbase + 16 * t + (lane & 15);
#pragma unroll
    for (int r = 0; r < 8; ++r)
      C[(size_t)(rbase + r) * N + col] = f2bf(acc[t][r]);
  }
}

// ---------------- embedding: circular conv (KE=3) + temporal -> bf16 --------
__global__ void embed_kernel(const float* __restrict__ x, const float* __restrict__ xm,
                             const float* __restrict__ ew, const float* __restrict__ tw,
                             bf16* __restrict__ h) {
  int idx = blockIdx.x * blockDim.x + threadIdx.x;
  if (idx >= MROWS * D_) return;
  const int d   = idx % D_;
  const int row = idx / D_;
  const int b = row / L_, l = row % L_;
  float acc = 0.f;
#pragma unroll
  for (int k = 0; k < KE_; ++k) {
    int li = l + k - 1;
    li = (li < 0) ? (L_ - 1) : ((li >= L_) ? 0 : li);
    const float* xr = x + ((size_t)b * L_ + li) * CIN_;
#pragma unroll
    for (int c = 0; c < CIN_; ++c)
      acc += xr[c] * ew[(d * CIN_ + c) * KE_ + k];
  }
  const float* xmr = xm + (size_t)row * NMARK_;
#pragma unroll
  for (int m = 0; m < NMARK_; ++m) acc += xmr[m] * tw[d * NMARK_ + m];
  h[idx] = f2bf(acc);
}

// ---------------- depthwise causal conv (K=4) + SiLU: xz(bf16) -> xc(f32) ----
__global__ void conv_silu_kernel(const bf16* __restrict__ xz, const float* __restrict__ cw,
                                 const float* __restrict__ cb, float* __restrict__ xc) {
  int idx = blockIdx.x * blockDim.x + threadIdx.x;
  if (idx >= MROWS * E_) return;
  const int e   = idx % E_;
  const int row = idx / E_;
  const int b = row / L_, l = row % L_;
  float acc = cb[e];
#pragma unroll
  for (int k = 0; k < K_; ++k) {
    const int li = l + k - (K_ - 1);
    if (li >= 0)
      acc += cw[e * K_ + k] * (float)xz[((size_t)b * L_ + li) * (2 * E_) + e];
  }
  xc[idx] = acc / (1.f + __expf(-acc));   // silu
}

// ---------------- x_proj: xdbl[row,f] = sum_e xc[row,e] * xpw[f,e] ----------
__global__ void xproj_kernel(const float* __restrict__ xc, const float* __restrict__ xpw,
                             float* __restrict__ xdbl) {
  int idx = blockIdx.x * blockDim.x + threadIdx.x;
  if (idx >= MROWS * F_) return;
  const int f   = idx % F_;
  const int row = idx / F_;
  const float* xr = xc + (size_t)row * E_;
  const float* wr = xpw + (size_t)f * E_;
  float acc = 0.f;
#pragma unroll 8
  for (int e = 0; e < E_; ++e) acc += xr[e] * wr[e];
  xdbl[idx] = acc;
}

// ============================================================================
// Chunked parallel selective scan (linear recurrence h_t = dA_t h + dBu_t).
// ============================================================================
__device__ __forceinline__ float softplus_f(float x) {
  return (x > 20.f) ? x : __logf(1.f + __expf(x));
}

__global__ __launch_bounds__(64)
void scan_pass1(const float* __restrict__ xc, const float* __restrict__ xdbl,
                const float* __restrict__ dtw, const float* __restrict__ dtb,
                const float* __restrict__ alog,
                float* __restrict__ cA, float* __restrict__ cC) {
  const int b = blockIdx.x;
  const int e = blockIdx.y * 64 + threadIdx.x;
  const int c = blockIdx.z;
  float a[N_], dw[R_], ap[N_], cr[N_];
#pragma unroll
  for (int n = 0; n < N_; ++n) { a[n] = -__expf(alog[e * N_ + n]); ap[n] = 1.f; cr[n] = 0.f; }
#pragma unroll
  for (int r = 0; r < R_; ++r) dw[r] = dtw[e * R_ + r];
  const float bias = dtb[e];
  for (int t = c * CL_; t < (c + 1) * CL_; ++t) {
    const size_t row = (size_t)b * L_ + t;
    const float* xd = xdbl + row * F_;
    float dtv = bias;
#pragma unroll
    for (int r = 0; r < R_; ++r) dtv += xd[r] * dw[r];
    const float delta = softplus_f(dtv);
    const float u = xc[row * E_ + e];
#pragma unroll
    for (int n = 0; n < N_; ++n) {
      const float dA = __expf(delta * a[n]);
      cr[n] = dA * cr[n] + (delta * xd[R_ + n]) * u;
      ap[n] *= dA;
    }
  }
  const size_t o = (((size_t)b * S_ + c) * E_ + e) * N_;
#pragma unroll
  for (int n = 0; n < N_; ++n) { cA[o + n] = ap[n]; cC[o + n] = cr[n]; }
}

__global__ __launch_bounds__(64)
void scan_pass2(const float* __restrict__ cA, const float* __restrict__ cC,
                float* __restrict__ hinit) {
  const int b = blockIdx.x;
  const int e = blockIdx.y * 64 + threadIdx.x;
  float h[N_];
#pragma unroll
  for (int n = 0; n < N_; ++n) h[n] = 0.f;
  for (int c = 0; c < S_; ++c) {
    const size_t o = (((size_t)b * S_ + c) * E_ + e) * N_;
#pragma unroll
    for (int n = 0; n < N_; ++n) hinit[o + n] = h[n];
#pragma unroll
    for (int n = 0; n < N_; ++n) h[n] = cA[o + n] * h[n] + cC[o + n];
  }
}

__global__ __launch_bounds__(64)
void scan_pass3(const float* __restrict__ xc, const float* __restrict__ xdbl,
                const bf16* __restrict__ xz, const float* __restrict__ dtw,
                const float* __restrict__ dtb, const float* __restrict__ alog,
                const float* __restrict__ dpar, const float* __restrict__ hinit,
                bf16* __restrict__ y) {
  const int b = blockIdx.x;
  const int e = blockIdx.y * 64 + threadIdx.x;
  const int c = blockIdx.z;
  float a[N_], dw[R_], h[N_];
  const size_t o = (((size_t)b * S_ + c) * E_ + e) * N_;
#pragma unroll
  for (int n = 0; n < N_; ++n) { a[n] = -__expf(alog[e * N_ + n]); h[n] = hinit[o + n]; }
#pragma unroll
  for (int r = 0; r < R_; ++r) dw[r] = dtw[e * R_ + r];
  const float bias = dtb[e];
  const float de   = dpar[e];
  for (int t = c * CL_; t < (c + 1) * CL_; ++t) {
    const size_t row = (size_t)b * L_ + t;
    const float* xd = xdbl + row * F_;
    float dtv = bias;
#pragma unroll
    for (int r = 0; r < R_; ++r) dtv += xd[r] * dw[r];
    const float delta = softplus_f(dtv);
    const float u = xc[row * E_ + e];
    float acc = 0.f;
#pragma unroll
    for (int n = 0; n < N_; ++n) {
      const float dA = __expf(delta * a[n]);
      h[n] = dA * h[n] + (delta * xd[R_ + n]) * u;       // Bm at [8..23]
      acc += h[n] * xd[R_ + N_ + n];                     // Cm at [24..39]
    }
    const float z  = (float)xz[row * (2 * E_) + E_ + e];
    const float sz = z / (1.f + __expf(-z));
    y[row * E_ + e] = f2bf((acc + u * de) * sz);
  }
}

// ---------------- head: out[b,c] = h[b,L-1,:] . proj_w[c,:] + proj_b[c] -----
__global__ void head_kernel(const bf16* __restrict__ h, const float* __restrict__ pw,
                            const float* __restrict__ pb, float* __restrict__ out) {
  const int t = threadIdx.x;
  if (t >= B_ * COUT_) return;
  const int b = t / COUT_, c = t % COUT_;
  const bf16* hr = h + ((size_t)b * L_ + (L_ - 1)) * D_;
  float acc = pb[c];
#pragma unroll
  for (int d = 0; d < D_; ++d) acc += (float)hr[d] * pw[c * D_ + d];
  out[t] = acc;
}

extern "C" void kernel_launch(void* const* d_in, const int* in_sizes, int n_in,
                              void* d_out, int out_size, void* d_ws, size_t ws_size,
                              hipStream_t stream) {
  (void)in_sizes; (void)n_in; (void)out_size; (void)ws_size;
  const float* x_enc      = (const float*)d_in[0];
  const float* x_mark_enc = (const float*)d_in[1];
  const float* emb_conv_w = (const float*)d_in[2];
  const float* temp_w     = (const float*)d_in[3];
  const float* in_proj_w  = (const float*)d_in[4];
  const float* conv_w     = (const float*)d_in[5];
  const float* conv_b     = (const float*)d_in[6];
  const float* x_proj_w   = (const float*)d_in[7];
  const float* dt_proj_w  = (const float*)d_in[8];
  const float* dt_proj_b  = (const float*)d_in[9];
  const float* A_log      = (const float*)d_in[10];
  const float* D_param    = (const float*)d_in[11];
  const float* out_proj_w = (const float*)d_in[12];
  const float* proj_w     = (const float*)d_in[13];
  const float* proj_b     = (const float*)d_in[14];

  // workspace layout (all offsets 16B-aligned)
  char* w = (char*)d_ws;
  bf16*  h    = (bf16*)w;  w += (size_t)MROWS * D_ * 2;           // 4 MB
  bf16*  xz   = (bf16*)w;  w += (size_t)MROWS * 2 * E_ * 2;       // 16 MB
  bf16*  y    = (bf16*)w;  w += (size_t)MROWS * E_ * 2;           // 8 MB
  float* xc   = (float*)w; w += (size_t)MROWS * E_ * 4;           // 16 MB
  float* xdbl = (float*)w; w += (size_t)MROWS * F_ * 4;           // 2.5 MB
  const size_t scansz = (size_t)B_ * E_ * S_ * N_;                // 1M elems
  float* cA    = (float*)w; w += scansz * 4;                      // 4 MB
  float* cC    = (float*)w; w += scansz * 4;                      // 4 MB
  float* hinit = (float*)w; w += scansz * 4;                      // 4 MB
  const int n_inw  = NL_ * 2 * E_ * D_;                           // 131072
  const int n_outw = NL_ * D_ * E_;                               // 65536
  bf16* wb_in  = (bf16*)w; w += (size_t)n_inw * 2;                // 256 KB
  bf16* wb_out = (bf16*)w; w += (size_t)n_outw * 2;               // 128 KB

  // one-shot weight conversion to bf16 (reused by every GEMM block)
  cvt_bf16_kernel<<<(n_inw + 255) / 256, 256, 0, stream>>>(in_proj_w, wb_in, n_inw);
  cvt_bf16_kernel<<<(n_outw + 255) / 256, 256, 0, stream>>>(out_proj_w, wb_out, n_outw);

  embed_kernel<<<(MROWS * D_ + 255) / 256, 256, 0, stream>>>(
      x_enc, x_mark_enc, emb_conv_w, temp_w, h);

  for (int i = 0; i < NL_; ++i) {
    const bf16*  inw  = wb_in      + (size_t)i * 2 * E_ * D_;
    const float* cw   = conv_w     + (size_t)i * E_ * K_;
    const float* cb   = conv_b     + (size_t)i * E_;
    const float* xpw  = x_proj_w   + (size_t)i * F_ * E_;
    const float* dtw  = dt_proj_w  + (size_t)i * E_ * R_;
    const float* dtb  = dt_proj_b  + (size_t)i * E_;
    const float* alog = A_log      + (size_t)i * E_ * N_;
    const float* dpar = D_param    + (size_t)i * E_;
    const bf16*  ow   = wb_out     + (size_t)i * D_ * E_;

    // xz = h @ in_w^T   (M=16384, N=512, K=128): 2 tiles/wave, 8 wmma/wave
    wmma_gemm_bf16<D_, 2><<<dim3(MROWS / 16, (2 * E_) / 256), 256, 0, stream>>>(
        h, inw, xz, 2 * E_);

    conv_silu_kernel<<<(MROWS * E_ + 255) / 256, 256, 0, stream>>>(xz, cw, cb, xc);

    xproj_kernel<<<(MROWS * F_ + 255) / 256, 256, 0, stream>>>(xc, xpw, xdbl);

    scan_pass1<<<dim3(B_, E_ / 64, S_), 64, 0, stream>>>(
        xc, xdbl, dtw, dtb, alog, cA, cC);
    scan_pass2<<<dim3(B_, E_ / 64), 64, 0, stream>>>(cA, cC, hinit);
    scan_pass3<<<dim3(B_, E_ / 64, S_), 64, 0, stream>>>(
        xc, xdbl, xz, dtw, dtb, alog, dpar, hinit, y);

    // h = y @ out_w^T   (M=16384, N=128, K=256): 1 tile/wave, 8 wmma/wave
    wmma_gemm_bf16<E_, 1><<<dim3(MROWS / 16, D_ / 128), 256, 0, stream>>>(
        y, ow, h, D_);
  }

  head_kernel<<<1, 32, 0, stream>>>(h, proj_w, proj_b, (float*)d_out);
}
